// DynamicWindowSparseAttention_61581241090091
// MI455X (gfx1250) — matrix operand order repaired and checked by reading, not compiled, not run on hardware
//
#include <hip/hip_runtime.h>
#include <math.h>

#define S_LEN  2048
#define DMODEL 1024
#define HEADS  16
#define HD     64
#define BATCH  2
#define MROWS  (BATCH * S_LEN)   // 4096
#define MAXT   10
#define MAXK   (MAXT * 16)       // 160

typedef __attribute__((ext_vector_type(16))) __bf16 v16bf;
typedef __attribute__((ext_vector_type(8)))  float  v8f;

static __device__ __forceinline__ v8f wmma_bf16(v16bf a, v16bf b, v8f c) {
  // D = A(16x32 bf16) x B(32x16 bf16) + C(16x16 f32)
  return __builtin_amdgcn_wmma_f32_16x16x32_bf16(false, a, false, b, (short)0, c, false, false);
}

// A-matrix fragment (16x32 bf16): lane 0-15 => rows M=0..15, K=0..15;
// lanes 16-31 => same rows, K=16..31. rowptr = start of this lane's row.
static __device__ __forceinline__ v16bf load_afrag(const __bf16* rowptr, int khalf) {
  union { v16bf v; uint4 u[2]; } r;
  r.u[0] = *(const uint4*)(rowptr + khalf * 16);
  r.u[1] = *(const uint4*)(rowptr + khalf * 16 + 8);
  return r.v;
}

// B-matrix fragment (32x16 bf16): lane 0-15 => col N=lane, K=0..15; lanes 16-31 => K=16..31.
static __device__ __forceinline__ v16bf load_bfrag_contig(const __bf16* p) {
  union { v16bf v; uint4 u[2]; } r;
  r.u[0] = *(const uint4*)(p);
  r.u[1] = *(const uint4*)(p + 8);
  return r.v;
}

// 32-bit LDS byte offset of a shared-memory object (generic -> AS3 -> int)
static __device__ __forceinline__ unsigned lds_addr_of(const void* p) {
  return (unsigned)(unsigned long long)(__attribute__((address_space(3))) const void*)p;
}

// CDNA5 async DMA: LDS[lds_addr] = MEM[gaddr], 16 bytes per lane, tracked by ASYNCcnt
static __device__ __forceinline__ void async_ld_b128(unsigned lds_addr, const __bf16* gaddr) {
  asm volatile("global_load_async_to_lds_b128 %0, %1, off"
               :: "v"(lds_addr), "v"(gaddr) : "memory");
}

// ---------------- dynamic window (step==0 path) ----------------
__global__ __launch_bounds__(256)
void window_kernel(const float* __restrict__ x, const float* __restrict__ alpha,
                   int* __restrict__ half_out) {
  __shared__ float red[256];
  int tid = threadIdx.x;
  float s = 0.f;
  for (int i = tid; i < BATCH * DMODEL; i += 256) {
    int b = i >> 10, d = i & (DMODEL - 1);
    size_t base = ((size_t)b * S_LEN + (S_LEN - 2)) * DMODEL + d;
    s += fabsf(x[base + DMODEL] - x[base]);
  }
  red[tid] = s;
  __syncthreads();
  for (int st = 128; st > 0; st >>= 1) {
    if (tid < st) red[tid] += red[tid + st];
    __syncthreads();
  }
  if (tid == 0) {
    float mean = red[0] / (float)(BATCH * DMODEL);
    size_t b0 = (size_t)(S_LEN - 2) * DMODEL;
    float ewma = fabsf(x[b0 + DMODEL] - x[b0]);     // fluct[0,0]
    float a = 1.f / (1.f + expf(-alpha[0]));
    float sig = a * mean + (1.f - a) * ewma;
    float sc = 1.f / (1.f + expf(-sig));
    int w = (int)(24.f + 96.f * sc);                // floor: positive
    if (w > 120) w = 120;
    if (w < 24) w = 24;
    half_out[0] = w >> 1;
  }
}

// ---------------- f32 -> bf16 cast, 8 elems/thread ----------------
__global__ __launch_bounds__(256)
void cast_f32_bf16(const float* __restrict__ in, __bf16* __restrict__ out, int n8) {
  int i = blockIdx.x * blockDim.x + threadIdx.x;
  if (i < n8) {
    const float4* p = (const float4*)in + (size_t)i * 2;
    float4 f0 = p[0], f1 = p[1];
    union { __bf16 e[8]; uint4 u; } o;
    o.e[0] = (__bf16)f0.x; o.e[1] = (__bf16)f0.y; o.e[2] = (__bf16)f0.z; o.e[3] = (__bf16)f0.w;
    o.e[4] = (__bf16)f1.x; o.e[5] = (__bf16)f1.y; o.e[6] = (__bf16)f1.z; o.e[7] = (__bf16)f1.w;
    *(uint4*)(out + (size_t)i * 8) = o.u;
  }
}

// ---------------- bf16 WMMA GEMM: C[M,N] = A[M,K] @ B[K,N] + bias ----------------
// M=4096, N=K=1024. 128x128 block tile, 8 waves, K-step 32.
// A tile: async global->LDS DMA, double-buffered. B tile: register pipeline + transpose.
template <bool OUTF>
__global__ __launch_bounds__(256)
void gemm_bf16_kernel(const __bf16* __restrict__ A, const __bf16* __restrict__ Bm,
                      const float* __restrict__ bias,
                      __bf16* __restrict__ outB, float* __restrict__ outF) {
  __shared__ __bf16 sA[2][128][48];   // [buf][m][k], stride 96B (16B aligned)
  __shared__ __bf16 sB[128][48];      // [n][k] (transposed: contiguous K per lane)
  const int tid  = threadIdx.x;
  const int wave = tid >> 5;
  const int lane = tid & 31;
  const int mBase = blockIdx.y * 128;
  const int nBase = blockIdx.x * 128;

  v8f acc[8] = {};

  const int arow = tid >> 1, acb = (tid & 1) * 16;   // A coop: 128x32, 32B/thread
  const int bkr  = tid >> 3, bcb = (tid & 7) * 16;   // B coop: 32x128
  const int fr    = lane & 15;
  const int khalf = lane >> 4;
  const int koff  = khalf * 16;

  const __bf16* gA = A + (size_t)(mBase + arow) * DMODEL + acb;
  const __bf16* gB = Bm + (size_t)bkr * DMODEL + nBase + bcb;
  const unsigned ldsA0 = lds_addr_of(&sA[0][arow][acb]);
  const unsigned ldsA1 = lds_addr_of(&sA[1][arow][acb]);

  // tile 0: A via async DMA, B via registers
  async_ld_b128(ldsA0,      gA);
  async_ld_b128(ldsA0 + 16, gA + 8);
  union { uint4 u[2]; __bf16 e[16]; } rb;
  rb.u[0] = *(const uint4*)(gB);
  rb.u[1] = *(const uint4*)(gB + 8);

  for (int it = 0; it < DMODEL / 32; ++it) {
    const int k0 = it * 32;
    const int buf = it & 1;

    // B registers -> LDS (transposed)
#pragma unroll
    for (int e = 0; e < 16; ++e) sB[bcb + e][bkr] = rb.e[e];

    // issue async DMA for next A tile into the other buffer
    if (it + 1 < DMODEL / 32) {
      const unsigned dst = buf ? ldsA0 : ldsA1;
      const __bf16* src = gA + (k0 + 32);
      async_ld_b128(dst,      src);
      async_ld_b128(dst + 16, src + 8);
      asm volatile("s_wait_asynccnt 0x2" ::: "memory");   // current tile done (in-order)
    } else {
      asm volatile("s_wait_asynccnt 0x0" ::: "memory");
    }
    __syncthreads();

    // next B tile global -> registers (overlaps WMMA burst)
    if (it + 1 < DMODEL / 32) {
      const __bf16* nB = gB + (size_t)(k0 + 32) * DMODEL;
      rb.u[0] = *(const uint4*)(nB);
      rb.u[1] = *(const uint4*)(nB + 8);
      if (it + 2 < DMODEL / 32)
        __builtin_prefetch(gB + (size_t)(k0 + 64) * DMODEL, 0, 1);
    }

    v16bf af = load_afrag(&sA[buf][wave * 16 + fr][0], khalf);
    v16bf bfr[8];
#pragma unroll
    for (int n = 0; n < 8; ++n) bfr[n] = load_bfrag_contig(&sB[n * 16 + fr][koff]);
#pragma unroll
    for (int n = 0; n < 8; ++n) acc[n] = wmma_bf16(af, bfr[n], acc[n]);
    __syncthreads();
  }

  float bv[8];
#pragma unroll
  for (int n = 0; n < 8; ++n) bv[n] = bias[nBase + n * 16 + fr];
#pragma unroll
  for (int n = 0; n < 8; ++n) {
#pragma unroll
    for (int v = 0; v < 8; ++v) {
      int M = mBase + wave * 16 + v + khalf * 8;
      int N = nBase + n * 16 + fr;
      float r = acc[n][v] + bv[n];
      if constexpr (OUTF) outF[(size_t)M * DMODEL + N] = r;
      else                outB[(size_t)M * DMODEL + N] = (__bf16)r;
    }
  }
}

// ---------------- banded attention + per-head W_o, one wave per 16-query tile ----------------
__global__ __launch_bounds__(32)
void attn_kernel(const __bf16* __restrict__ Q, const __bf16* __restrict__ K,
                 const __bf16* __restrict__ V, const __bf16* __restrict__ Wo,
                 __bf16* __restrict__ out, const int* __restrict__ half_p) {
  __shared__ float  s_sc[16][MAXK];
  __shared__ __bf16 s_p[16][MAXK];
  __shared__ __bf16 s_ctx[16][HD];
  __shared__ __bf16 sVT[HD][48];   // [dim][key-in-chunk], stride 96B
  __shared__ __bf16 sWoT[HD][80];  // [k_out][d] transposed, stride 160B

  const int lane = threadIdx.x;
  const int q0 = blockIdx.x * 16;
  const int bh = blockIdx.y;
  const int b = bh >> 4, h = bh & 15;

  const int whalf = half_p[0];
  int jlo = q0 - whalf;            if (jlo < 0) jlo = 0;
  int jhi = q0 + 15 + whalf;       if (jhi > S_LEN - 1) jhi = S_LEN - 1;
  const int t0 = jlo >> 4, t1 = jhi >> 4;
  const int nt = t1 - t0 + 1;      // <= 9
  const int nk = nt * 16;
  const int nchunk = (nt + 1) >> 1;

  const int rowq  = lane & 15;
  const int khalf = lane >> 4;
  const int bn    = lane & 15;
  const int koff  = khalf * 16;

  // stage W_o[h] transposed into LDS: sWoT[k][d] = Wo[h][d][k]
  {
    const __bf16* wh = Wo + (size_t)h * HD * HD;
#pragma unroll
    for (int rr = 0; rr < 2; ++rr) {
      int d = lane + rr * 32;
      union { uint4 u[8]; __bf16 e[64]; } row;
#pragma unroll
      for (int i = 0; i < 8; ++i) row.u[i] = ((const uint4*)(wh + (size_t)d * HD))[i];
#pragma unroll
      for (int k = 0; k < 64; ++k) sWoT[k][d] = row.e[k];
    }
  }

  // Q fragments (16x64 split into two 16x32 chunks)
  const __bf16* qrow = Q + ((size_t)(b * S_LEN + q0 + rowq)) * DMODEL + h * HD;
  v16bf qa0 = load_afrag(qrow, khalf);
  v16bf qa1 = load_afrag(qrow + 32, khalf);

  // scores = Q K^T * 1/sqrt(hd), band mask
  for (int t = t0; t <= t1; ++t) {
    const __bf16* kp = K + ((size_t)(b * S_LEN + t * 16 + bn)) * DMODEL + h * HD;
    v16bf kb0 = load_bfrag_contig(kp + koff);
    v16bf kb1 = load_bfrag_contig(kp + 32 + koff);
    v8f acc = {};
    acc = wmma_bf16(qa0, kb0, acc);
    acc = wmma_bf16(qa1, kb1, acc);
#pragma unroll
    for (int v = 0; v < 8; ++v) {
      int M = v + khalf * 8;
      int i = q0 + M, j = t * 16 + bn;
      float sv = acc[v] * 0.125f;              // 1/sqrt(64)
      int d = j - i; if (d < 0) d = -d;
      if (d > whalf) sv = -1e30f;
      s_sc[M][(t - t0) * 16 + bn] = sv;
    }
  }
  __syncthreads();

  // row softmax (16 lanes, one row each), write normalized bf16 P; zero-pad to chunk
  if (lane < 16) {
    int r = lane;
    float m = -3.4e38f;
    for (int c = 0; c < nk; ++c) m = fmaxf(m, s_sc[r][c]);
    float sum = 0.f;
    for (int c = 0; c < nk; ++c) { float e = __expf(s_sc[r][c] - m); s_sc[r][c] = e; sum += e; }
    float inv = 1.f / sum;
    for (int c = 0; c < nk; ++c) s_p[r][c] = (__bf16)(s_sc[r][c] * inv);
    for (int c = nk; c < nchunk * 32; ++c) s_p[r][c] = (__bf16)0.0f;
  }
  __syncthreads();

  // ctx = P @ V  (K-dim = keys, 32 per WMMA); V chunk staged through LDS transposed
  v8f accc[4] = {};
  for (int kc = 0; kc < nchunk; ++kc) {
    {
      int key = t0 * 16 + kc * 32 + lane;
      if (key > S_LEN - 1) key = S_LEN - 1;        // OOB keys have P==0
      const __bf16* vrow = V + ((size_t)(b * S_LEN + key)) * DMODEL + h * HD;
      union { uint4 u[8]; __bf16 e[64]; } row;
#pragma unroll
      for (int i = 0; i < 8; ++i) row.u[i] = ((const uint4*)vrow)[i];
#pragma unroll
      for (int d = 0; d < 64; ++d) sVT[d][lane] = row.e[d];
    }
    __syncthreads();
    v16bf pa = load_afrag(&s_p[rowq][kc * 32], khalf);
    v16bf vb[4];
#pragma unroll
    for (int n = 0; n < 4; ++n) vb[n] = load_bfrag_contig(&sVT[n * 16 + bn][koff]);
#pragma unroll
    for (int n = 0; n < 4; ++n) accc[n] = wmma_bf16(pa, vb[n], accc[n]);
    __syncthreads();
  }
#pragma unroll
  for (int n = 0; n < 4; ++n)
#pragma unroll
    for (int v = 0; v < 8; ++v)
      s_ctx[v + khalf * 8][n * 16 + bn] = (__bf16)accc[n][v];
  __syncthreads();

  // out_head = ctx @ W_o[h]  (64x64), W_o fragments from LDS (contiguous K)
  v16bf ca0 = load_afrag(&s_ctx[rowq][0], khalf);
  v16bf ca1 = load_afrag(&s_ctx[rowq][32], khalf);
  v8f acco[4] = {};
#pragma unroll
  for (int n = 0; n < 4; ++n) {
    v16bf w0 = load_bfrag_contig(&sWoT[n * 16 + bn][koff]);
    v16bf w1 = load_bfrag_contig(&sWoT[n * 16 + bn][32 + koff]);
    acco[n] = wmma_bf16(ca0, w0, acco[n]);
    acco[n] = wmma_bf16(ca1, w1, acco[n]);
  }
#pragma unroll
  for (int n = 0; n < 4; ++n)
#pragma unroll
    for (int v = 0; v < 8; ++v) {
      int M = v + khalf * 8;
      out[((size_t)(b * S_LEN + q0 + M)) * DMODEL + h * HD + n * 16 + bn] = (__bf16)acco[n][v];
    }
}

// ---------------- launch ----------------
extern "C" void kernel_launch(void* const* d_in, const int* in_sizes, int n_in,
                              void* d_out, int out_size, void* d_ws, size_t ws_size,
                              hipStream_t stream) {
  (void)in_sizes; (void)n_in; (void)out_size; (void)ws_size;
  const float* x     = (const float*)d_in[0];
  const float* alpha = (const float*)d_in[1];
  const float* Wq    = (const float*)d_in[2];
  const float* bq    = (const float*)d_in[3];
  const float* Wk    = (const float*)d_in[4];
  const float* bk    = (const float*)d_in[5];
  const float* Wv    = (const float*)d_in[6];
  const float* bv    = (const float*)d_in[7];
  const float* Wo    = (const float*)d_in[8];
  const float* Wfc   = (const float*)d_in[9];
  const float* bfc   = (const float*)d_in[10];
  float* out = (float*)d_out;

  char* p = (char*)d_ws;
  int* half_i = (int*)p;            p += 256;
  __bf16* xb   = (__bf16*)p;        p += (size_t)MROWS * DMODEL * 2;
  __bf16* wqb  = (__bf16*)p;        p += (size_t)DMODEL * DMODEL * 2;
  __bf16* wkb  = (__bf16*)p;        p += (size_t)DMODEL * DMODEL * 2;
  __bf16* wvb  = (__bf16*)p;        p += (size_t)DMODEL * DMODEL * 2;
  __bf16* wfcb = (__bf16*)p;        p += (size_t)DMODEL * DMODEL * 2;
  __bf16* wob  = (__bf16*)p;        p += (size_t)HEADS * HD * HD * 2;
  __bf16* Qb   = (__bf16*)p;        p += (size_t)MROWS * DMODEL * 2;
  __bf16* Kb   = (__bf16*)p;        p += (size_t)MROWS * DMODEL * 2;
  __bf16* Vb   = (__bf16*)p;        p += (size_t)MROWS * DMODEL * 2;
  __bf16* Cb   = (__bf16*)p;        p += (size_t)MROWS * DMODEL * 2;

  window_kernel<<<1, 256, 0, stream>>>(x, alpha, half_i);

  const int nx8 = (MROWS * DMODEL) / 8;
  const int nw8 = (DMODEL * DMODEL) / 8;
  const int no8 = (HEADS * HD * HD) / 8;
  cast_f32_bf16<<<(nx8 + 255) / 256, 256, 0, stream>>>(x,   xb,   nx8);
  cast_f32_bf16<<<(nw8 + 255) / 256, 256, 0, stream>>>(Wq,  wqb,  nw8);
  cast_f32_bf16<<<(nw8 + 255) / 256, 256, 0, stream>>>(Wk,  wkb,  nw8);
  cast_f32_bf16<<<(nw8 + 255) / 256, 256, 0, stream>>>(Wv,  wvb,  nw8);
  cast_f32_bf16<<<(nw8 + 255) / 256, 256, 0, stream>>>(Wfc, wfcb, nw8);
  cast_f32_bf16<<<(no8 + 255) / 256, 256, 0, stream>>>(Wo,  wob,  no8);

  dim3 ggrid(DMODEL / 128, MROWS / 128);
  gemm_bf16_kernel<false><<<ggrid, 256, 0, stream>>>(xb, wqb, bq, Qb, nullptr);
  gemm_bf16_kernel<false><<<ggrid, 256, 0, stream>>>(xb, wkb, bk, Kb, nullptr);
  gemm_bf16_kernel<false><<<ggrid, 256, 0, stream>>>(xb, wvb, bv, Vb, nullptr);

  attn_kernel<<<dim3(S_LEN / 16, BATCH * HEADS), 32, 0, stream>>>(Qb, Kb, Vb, wob, Cb, half_i);

  gemm_bf16_kernel<true><<<ggrid, 256, 0, stream>>>(Cb, wfcb, bfc, nullptr, out);
}
